// LagunaAttention_27462020891147
// MI455X (gfx1250) — matrix operand order repaired and checked by reading, not compiled, not run on hardware
//
#include <hip/hip_runtime.h>
#include <math.h>

#define T_TOK 4096
#define HID   2048
#define NH    16
#define NKV   4
#define HD    128
#define QKV_OUT 3072   // (16 + 2*4) * 128

typedef __attribute__((ext_vector_type(16))) __bf16 v16bf;
typedef __attribute__((ext_vector_type(8)))  __bf16 v8bf;
typedef __attribute__((ext_vector_type(8)))  float  v8f;
typedef __attribute__((ext_vector_type(4)))  int    v4i;

__device__ __forceinline__ v16bf cat8(v8bf lo, v8bf hi) {
  return __builtin_shufflevector(lo, hi, 0,1,2,3,4,5,6,7,8,9,10,11,12,13,14,15);
}
__device__ __forceinline__ v8f wmma_bf16(v16bf a, v16bf b, v8f c) {
  // D = A(16x32 bf16) * B(32x16 bf16) + C(16x16 f32)
  return __builtin_amdgcn_wmma_f32_16x16x32_bf16(false, a, false, b, (short)0, c, false, false);
}

// ---------------------------------------------------------------------------
// Async global->LDS copy (gfx1250 GLOBAL_LOAD_ASYNC_TO_LDS_B128, ASYNCcnt),
// with a synchronous fallback if the builtins are unavailable.
// Builtin signature (probed from diagnostic): (v4i AS1*, v4i AS3*, Ii, Ii).
// ---------------------------------------------------------------------------
#if defined(__AMDGCN__) &&                                           \
    __has_builtin(__builtin_amdgcn_global_load_async_to_lds_b128) && \
    __has_builtin(__builtin_amdgcn_s_wait_asynccnt)
#define HAS_ASYNC_LDS 1
typedef __attribute__((address_space(1))) v4i* g_v4i_ptr;
typedef __attribute__((address_space(3))) v4i* l_v4i_ptr;
__device__ __forceinline__ void async_copy_b128(const void* g, void* l) {
  __builtin_amdgcn_global_load_async_to_lds_b128(
      (g_v4i_ptr)(void*)g, (l_v4i_ptr)l, /*imm offset*/0, /*cpol*/0);
}
__device__ __forceinline__ void wait_async_copies() {
  __builtin_amdgcn_s_wait_asynccnt(0);
}
#else
#define HAS_ASYNC_LDS 0
__device__ __forceinline__ void async_copy_b128(const void* g, void* l) {
  *(v8bf*)l = *(const v8bf*)g;
}
__device__ __forceinline__ void wait_async_copies() {}
#endif

// ---------------------------------------------------------------------------
// fp32 -> bf16 conversion (sources are single-use: non-temporal loads)
// ---------------------------------------------------------------------------
__global__ __launch_bounds__(256) void cvt_f32_bf16_kernel(
    const float* __restrict__ src, __bf16* __restrict__ dst, int n) {
  int i = blockIdx.x * 256 + threadIdx.x;
  if (i < n) dst[i] = (__bf16)__builtin_nontemporal_load(src + i);
}

// ---------------------------------------------------------------------------
// bf16 WMMA GEMM: C[M,N] f32 = A[M,K] bf16 (row major) * B[K,N] bf16 (row major)
// Block tile 128x64, BK=32, 256 threads = 8 waves arranged 4x2, wave tile 32x32.
// A tile staged via async global->LDS DMA; B tile staged transposed (sync).
// ---------------------------------------------------------------------------
#define SA_STRIDE 40   // 32 + 8 pad elems (80B rows -> 16B aligned b128 chunks)
#define SB_STRIDE 40

__global__ __launch_bounds__(256) void gemm_bf16_kernel(
    const __bf16* __restrict__ A, const __bf16* __restrict__ B,
    float* __restrict__ C, int M, int N, int K) {
  (void)M;
  __shared__ __bf16 sA[128 * SA_STRIDE];   // [m][k] row-major
  __shared__ __bf16 sBt[64 * SB_STRIDE];   // [n][k] (B transposed)

  const int tid  = threadIdx.x;
  const int wave = tid >> 5, lane = tid & 31;
  const int half = lane >> 4, l16 = lane & 15;
  const int wrow = wave >> 1, wcol = wave & 1;      // 4 x 2 waves
  const int bm = blockIdx.y * 128, bn = blockIdx.x * 64;

  v8f acc[2][2] = {};

  for (int k0 = 0; k0 < K; k0 += 32) {
    // stage A tile (128x32): async b128 global->LDS, no VGPR round-trip
    {
      int r = tid >> 2;            // 0..63
      int c = (tid & 3) * 8;       // 0,8,16,24
      async_copy_b128(&A[(size_t)(bm + r) * K + k0 + c],
                      &sA[r * SA_STRIDE + c]);
      async_copy_b128(&A[(size_t)(bm + r + 64) * K + k0 + c],
                      &sA[(r + 64) * SA_STRIDE + c]);
    }
    // stage B tile (32x64) transposed into sBt[n][k] (sync: scatter transpose)
    {
      int kr = tid >> 3;           // 0..31
      int nc = (tid & 7) * 8;      // 0..56
      v8bf tmp = *(const v8bf*)&B[(size_t)(k0 + kr) * N + bn + nc];
      #pragma unroll
      for (int j = 0; j < 8; ++j) sBt[(nc + j) * SB_STRIDE + kr] = tmp[j];
    }
    if (k0 + 32 < K) {
      __builtin_prefetch(&A[(size_t)(bm + (tid >> 2)) * K + k0 + 32], 0, 1);
      __builtin_prefetch(&B[(size_t)(k0 + 32 + (tid >> 3)) * N + bn], 0, 1);
    }
    wait_async_copies();       // ASYNCcnt -> 0 before signaling the barrier
    __syncthreads();

    // gather fragments (each = 2x ds_load_b128 thanks to layout match)
    v16bf afrag[2], bfrag[2];
    #pragma unroll
    for (int mi = 0; mi < 2; ++mi) {
      int row = wrow * 32 + mi * 16 + l16;        // A: M = lane%16
      v8bf lo = *(const v8bf*)&sA[row * SA_STRIDE + half * 8];        // K = 8h..8h+7
      v8bf hi = *(const v8bf*)&sA[row * SA_STRIDE + 16 + half * 8];   // K = 16+8h..
      afrag[mi] = cat8(lo, hi);
    }
    #pragma unroll
    for (int ni = 0; ni < 2; ++ni) {
      int col = wcol * 32 + ni * 16 + l16;        // B: N = lane%16
      v8bf lo = *(const v8bf*)&sBt[col * SB_STRIDE + half * 8];       // K = 8h..8h+7
      v8bf hi = *(const v8bf*)&sBt[col * SB_STRIDE + 16 + half * 8];  // K = 16+8h..
      bfrag[ni] = cat8(lo, hi);
    }
    #pragma unroll
    for (int mi = 0; mi < 2; ++mi)
      #pragma unroll
      for (int ni = 0; ni < 2; ++ni)
        acc[mi][ni] = wmma_bf16(afrag[mi], bfrag[ni], acc[mi][ni]);
    __syncthreads();
  }

  // C layout: VGPR r, lanes0-15 -> M=r, lanes16-31 -> M=r+8; N = lane%16
  #pragma unroll
  for (int mi = 0; mi < 2; ++mi)
    #pragma unroll
    for (int ni = 0; ni < 2; ++ni) {
      int col = bn + wcol * 32 + ni * 16 + l16;
      #pragma unroll
      for (int r = 0; r < 8; ++r) {
        int row = bm + wrow * 32 + mi * 16 + r + 8 * half;
        C[(size_t)row * N + col] = acc[mi][ni][r];
      }
    }
}

// ---------------------------------------------------------------------------
// Per-token: split qkv, RMS-norm q/k, NeoX RoPE q/k, bf16 pack; softplus gate.
// One block per token, 8 waves. Wave handles heads {w, w+8, w+16} of 24.
// Lane holds d = lane*4..lane*4+3; rotary halves live in lanes 0-15 / 16-31.
// ---------------------------------------------------------------------------
__global__ __launch_bounds__(256) void qkv_post_kernel(
    const float* __restrict__ qkv, const float* __restrict__ hidden,
    const float* __restrict__ w_g, const int* __restrict__ positions,
    const float* __restrict__ qnw, const float* __restrict__ knw,
    __bf16* __restrict__ Qb, __bf16* __restrict__ Kb, __bf16* __restrict__ Vb,
    float* __restrict__ gate) {
  const int t = blockIdx.x;
  const int tid = threadIdx.x, wave = tid >> 5, lane = tid & 31;
  const float pos = (float)positions[t];
  const float* row = qkv + (size_t)t * QKV_OUT;
  const int d0 = lane * 4;

  for (int hh = wave; hh < 24; hh += 8) {
    const float* hp = row + hh * HD;
    float x[4];
    #pragma unroll
    for (int i = 0; i < 4; ++i) x[i] = hp[d0 + i];

    float ss = x[0]*x[0] + x[1]*x[1] + x[2]*x[2] + x[3]*x[3];
    #pragma unroll
    for (int m = 1; m <= 16; m <<= 1) ss += __shfl_xor(ss, m, 32);
    const float inv = rsqrtf(ss * (1.0f / 128.0f) + 1e-6f);

    if (hh < 20) {  // q and k heads: norm + rope
      const float* nw = (hh < 16) ? qnw : knw;
      #pragma unroll
      for (int i = 0; i < 4; ++i) x[i] = x[i] * inv * nw[d0 + i];
      #pragma unroll
      for (int i = 0; i < 4; ++i) {
        int f = (lane & 15) * 4 + i;                 // rotary freq idx 0..63
        float ang = pos * __expf(-(float)f * 0.14391157f);  // ln(1e4)/64
        float sn, cs; __sincosf(ang, &sn, &cs);
        float p = __shfl_xor(x[i], 16, 32);          // partner half
        x[i] = (lane < 16) ? (x[i] * cs - p * sn) : (x[i] * cs + p * sn);
      }
    }
    #pragma unroll
    for (int i = 0; i < 4; ++i) {
      __bf16 b = (__bf16)x[i];
      int d = d0 + i;
      if (hh < 16)      Qb[((size_t)hh * T_TOK + t) * HD + d] = b;
      else if (hh < 20) Kb[((size_t)(hh - 16) * T_TOK + t) * HD + d] = b;
      else              Vb[((size_t)(hh - 20) * T_TOK + t) * HD + d] = b;
    }
  }

  // gate[t][h] = softplus(hidden[t,:] . w_g[:,h])
  for (int h = wave; h < NH; h += 8) {
    float acc = 0.f;
    for (int i = lane; i < HID; i += 32)
      acc += hidden[(size_t)t * HID + i] * w_g[(size_t)i * NH + h];
    #pragma unroll
    for (int m = 1; m <= 16; m <<= 1) acc += __shfl_xor(acc, m, 32);
    if (lane == 0) {
      float g = (acc > 20.f) ? acc : log1pf(__expf(acc));
      gate[(size_t)t * NH + h] = g;
    }
  }
}

// ---------------------------------------------------------------------------
// Flash attention: grid (qblock, head), 128 threads = 4 waves.
// Wave owns 16 q rows; KV processed in 64-row blocks with online softmax.
// QK^T: Q frags + K frags straight from global (K row-major == B layout).
// P@V : V staged transposed in LDS, P restriped via per-wave LDS.
// ---------------------------------------------------------------------------
#define SVT_STRIDE 72  // 64 + 8 pad (144B rows)
#define SP_STRIDE  72

__global__ __launch_bounds__(128) void attn_kernel(
    const __bf16* __restrict__ Q, const __bf16* __restrict__ Kc,
    const __bf16* __restrict__ V, const float* __restrict__ gate,
    __bf16* __restrict__ Out) {
  const int qb = blockIdx.x;          // 64-row q block
  const int h  = blockIdx.y;
  const int hkv = h >> 2;             // GQA: rep = 4
  const int tid = threadIdx.x, wave = tid >> 5, lane = tid & 31;
  const int half = lane >> 4, l16 = lane & 15;
  const int qrow0 = qb * 64 + wave * 16;

  __shared__ __bf16 sVt[128 * SVT_STRIDE];      // [d][kv]
  __shared__ __bf16 sP[4][16 * SP_STRIDE];      // per wave [m][kv]

  // Q fragments for this wave's 16 rows (A layout: M = lane%16)
  v16bf qfrag[4];
  {
    const __bf16* Qrow = Q + ((size_t)h * T_TOK + qrow0 + l16) * HD;
    #pragma unroll
    for (int dk = 0; dk < 4; ++dk) {
      v8bf lo = *(const v8bf*)(Qrow + dk * 32 + 8 * half);
      v8bf hi = *(const v8bf*)(Qrow + dk * 32 + 16 + 8 * half);
      qfrag[dk] = cat8(lo, hi);
    }
  }

  v8f o[8] = {};
  float mrow[8], lrow[8];
  #pragma unroll
  for (int r = 0; r < 8; ++r) { mrow[r] = -__builtin_inff(); lrow[r] = 0.f; }

  for (int kb = 0; kb <= qb; ++kb) {
    const int kv0 = kb * 64;
    __syncthreads();  // protect previous iteration's sVt reads
    // stage V block transposed: sVt[d][kv]
    for (int it = tid; it < 64 * 16; it += 128) {
      int r  = it >> 4;            // kv row 0..63
      int cg = (it & 15) * 8;      // d group
      v8bf tmp = *(const v8bf*)&V[((size_t)hkv * T_TOK + kv0 + r) * HD + cg];
      #pragma unroll
      for (int j = 0; j < 8; ++j) sVt[(cg + j) * SVT_STRIDE + r] = tmp[j];
    }
    __syncthreads();

    // S = scale * Q K^T  (16 WMMAs)
    v8f s[4] = {};
    #pragma unroll
    for (int nt = 0; nt < 4; ++nt) {
      const __bf16* Krow = Kc + ((size_t)hkv * T_TOK + kv0 + nt * 16 + l16) * HD;
      #pragma unroll
      for (int dk = 0; dk < 4; ++dk) {
        v8bf lo = *(const v8bf*)(Krow + dk * 32 + 8 * half);
        v8bf hi = *(const v8bf*)(Krow + dk * 32 + 16 + 8 * half);
        s[nt] = wmma_bf16(qfrag[dk], cat8(lo, hi), s[nt]);
      }
    }
    // scale + causal mask (diagonal block only)
    #pragma unroll
    for (int nt = 0; nt < 4; ++nt)
      #pragma unroll
      for (int r = 0; r < 8; ++r) {
        float v = s[nt][r] * 0.08838834764831845f;   // 1/sqrt(128)
        if (kb == qb && (kv0 + nt * 16 + l16) > (qrow0 + r + 8 * half))
          v = -__builtin_inff();
        s[nt][r] = v;
      }

    // online softmax per row (rows r+8*half live on this lane's half)
    #pragma unroll
    for (int r = 0; r < 8; ++r) {
      float mx = -__builtin_inff();
      #pragma unroll
      for (int nt = 0; nt < 4; ++nt) mx = fmaxf(mx, s[nt][r]);
      #pragma unroll
      for (int m = 1; m <= 8; m <<= 1) mx = fmaxf(mx, __shfl_xor(mx, m, 32));
      float mnew  = fmaxf(mrow[r], mx);
      float alpha = __expf(mrow[r] - mnew);
      float rs = 0.f;
      #pragma unroll
      for (int nt = 0; nt < 4; ++nt) {
        float p = __expf(s[nt][r] - mnew);
        rs += p;
        sP[wave][(r + 8 * half) * SP_STRIDE + nt * 16 + l16] = (__bf16)p;
      }
      #pragma unroll
      for (int m = 1; m <= 8; m <<= 1) rs += __shfl_xor(rs, m, 32);
      lrow[r] = lrow[r] * alpha + rs;
      mrow[r] = mnew;
      #pragma unroll
      for (int dt = 0; dt < 8; ++dt) o[dt][r] *= alpha;
    }

    // O += P @ V  (16 WMMAs; P refetched in A layout, V^T rows give B layout)
    #pragma unroll
    for (int ks = 0; ks < 2; ++ks) {
      v8bf plo = *(const v8bf*)&sP[wave][l16 * SP_STRIDE + ks * 32 + 8 * half];
      v8bf phi = *(const v8bf*)&sP[wave][l16 * SP_STRIDE + ks * 32 + 16 + 8 * half];
      v16bf pfrag = cat8(plo, phi);
      #pragma unroll
      for (int dt = 0; dt < 8; ++dt) {
        v8bf vlo = *(const v8bf*)&sVt[(dt * 16 + l16) * SVT_STRIDE + ks * 32 + 8 * half];
        v8bf vhi = *(const v8bf*)&sVt[(dt * 16 + l16) * SVT_STRIDE + ks * 32 + 16 + 8 * half];
        o[dt] = wmma_bf16(pfrag, cat8(vlo, vhi), o[dt]);
      }
    }
  }

  // normalize, gate, store bf16 [t][NH*HD]
  #pragma unroll
  for (int r = 0; r < 8; ++r) {
    int row = qrow0 + r + 8 * half;
    float g = gate[(size_t)row * NH + h];
    float f = g / lrow[r];
    #pragma unroll
    for (int dt = 0; dt < 8; ++dt)
      Out[(size_t)row * (NH * HD) + h * HD + dt * 16 + l16] = (__bf16)(o[dt][r] * f);
  }
}

// ---------------------------------------------------------------------------
extern "C" void kernel_launch(void* const* d_in, const int* in_sizes, int n_in,
                              void* d_out, int out_size, void* d_ws, size_t ws_size,
                              hipStream_t stream) {
  (void)in_sizes; (void)n_in; (void)out_size; (void)ws_size;
  const float* hidden    = (const float*)d_in[0];
  const int*   positions = (const int*)d_in[1];
  const float* w_qkv     = (const float*)d_in[2];
  const float* w_o       = (const float*)d_in[3];
  const float* w_g       = (const float*)d_in[4];
  const float* qnw       = (const float*)d_in[5];
  const float* knw       = (const float*)d_in[6];
  float* out = (float*)d_out;

  char* ws = (char*)d_ws;
  size_t off = 0;
  auto take = [&](size_t bytes) -> char* {
    char* p = ws + off;
    off += (bytes + 255) & ~(size_t)255;
    return p;
  };
  __bf16* Xbf   = (__bf16*)take((size_t)T_TOK * HID * 2);
  __bf16* Wqkvb = (__bf16*)take((size_t)HID * QKV_OUT * 2);
  __bf16* Wob   = (__bf16*)take((size_t)HID * HID * 2);
  float*  qkv32 = (float*) take((size_t)T_TOK * QKV_OUT * 4);
  __bf16* Qb    = (__bf16*)take((size_t)NH  * T_TOK * HD * 2);
  __bf16* Kb    = (__bf16*)take((size_t)NKV * T_TOK * HD * 2);
  __bf16* Vb    = (__bf16*)take((size_t)NKV * T_TOK * HD * 2);
  float*  gateb = (float*) take((size_t)T_TOK * NH * 4);
  __bf16* Ab    = (__bf16*)take((size_t)T_TOK * HID * 2);

  int n1 = T_TOK * HID, n2 = HID * QKV_OUT, n3 = HID * HID;
  cvt_f32_bf16_kernel<<<(n1 + 255) / 256, 256, 0, stream>>>(hidden, Xbf, n1);
  cvt_f32_bf16_kernel<<<(n2 + 255) / 256, 256, 0, stream>>>(w_qkv, Wqkvb, n2);
  cvt_f32_bf16_kernel<<<(n3 + 255) / 256, 256, 0, stream>>>(w_o, Wob, n3);

  gemm_bf16_kernel<<<dim3(QKV_OUT / 64, T_TOK / 128), 256, 0, stream>>>(
      Xbf, Wqkvb, qkv32, T_TOK, QKV_OUT, HID);

  qkv_post_kernel<<<T_TOK, 256, 0, stream>>>(
      qkv32, hidden, w_g, positions, qnw, knw, Qb, Kb, Vb, gateb);

  attn_kernel<<<dim3(T_TOK / 64, NH), 128, 0, stream>>>(Qb, Kb, Vb, gateb, Ab);

  gemm_bf16_kernel<<<dim3(HID / 64, T_TOK / 128), 256, 0, stream>>>(
      Ab, Wob, out, T_TOK, HID, HID);
}
